// GINConvNet_12240656794168
// MI455X (gfx1250) — matrix operand order repaired and checked by reading, not compiled, not run on hardware
//
#include <hip/hip_runtime.h>
#include <hip/hip_bf16.h>

// ---------------------------------------------------------------------------
// GINConvNet (DeepDDS) forward for MI455X / gfx1250, wave32 + WMMA f16.
// ---------------------------------------------------------------------------

#define NN 250000
#define EE 1000000
#define BB 8192
#define FXD 78
#define DIM 32
#define BN_EPS 1e-5f

typedef __attribute__((ext_vector_type(16))) _Float16 v16h;
typedef __attribute__((ext_vector_type(8)))  _Float16 v8h;
typedef __attribute__((ext_vector_type(8)))  float    v8f;

#define AS_STRIDE 40   // halves; 80 B row stride: 16B-aligned, 20-bank spread
#define BT_STRIDE 40   // halves; transposed B tile BsT[n][k]

// ---------------------------------------------------------------------------
// Generic GEMM: C[m*ldc+n] = act(A[m*lda + k] * B[k*N + n] + bias[n])
// Block tile 128(M) x 32(N); 8 waves, each wave owns a 16x32 strip.
// WMMA f32 <- f16 x f16, K stepped by 32 (zero-padded remainder in LDS).
// ---------------------------------------------------------------------------
__global__ __launch_bounds__(256)
void gemm_wmma_f16(const float* __restrict__ A, const float* __restrict__ Bm,
                   const float* __restrict__ bias, float* __restrict__ C,
                   int M, int N, int K, int lda, int ldc, int relu)
{
    __shared__ __align__(16) _Float16 As[128 * AS_STRIDE];
    __shared__ __align__(16) _Float16 BsT[32 * BT_STRIDE];

    const int t    = threadIdx.x;
    const int lane = t & 31;
    const int wave = t >> 5;            // 0..7
    const int m0   = blockIdx.x * 128;
    const int n0   = blockIdx.y * 32;

    const bool fullM = (m0 + 128 <= M);
    const bool fullN = (n0 + 32 <= N);

    v8f acc0 = {};
    v8f acc1 = {};

    const int ksteps = (K + 31) >> 5;
    for (int ks = 0; ks < ksteps; ++ks) {
        const int kb = ks << 5;
        const bool fullK = (kb + 32 <= K);

        // ---- stage A tile: 128 rows x 32 k; 2 threads/row, 16 halves each
        {
            const int  r  = t >> 1;
            const int  kh = (t & 1) << 4;          // 0 or 16
            const long gr = (long)m0 + r;
            _Float16* dst = &As[r * AS_STRIDE + kh];
            if (fullM && fullK) {
                // uniform fast path: 8x float2 (8B-aligned: lda even, kh even)
                const float* ap = A + gr * (long)lda + kb + kh;
                #pragma unroll
                for (int j = 0; j < 8; ++j) {
                    const float2 v = *(const float2*)(ap + 2 * j);
                    union { _Float16 h[2]; unsigned u; } pk;
                    pk.h[0] = (_Float16)v.x;
                    pk.h[1] = (_Float16)v.y;
                    *(unsigned*)(dst + 2 * j) = pk.u;   // ds_store_b32
                }
            } else {
                // ragged tile: clamped-address unconditional loads + select 0
                // (no EXEC divergence, loads batched in flight)
                const bool rowok = (gr < M);
                const long grc   = rowok ? gr : (long)(M - 1);
                const float* ap  = A + grc * (long)lda;
                #pragma unroll
                for (int j = 0; j < 16; ++j) {
                    const int k  = kb + kh + j;
                    const int kc = (k < K) ? k : (K - 1);
                    float v = ap[kc];
                    v = (rowok && (k < K)) ? v : 0.0f;
                    dst[j] = (_Float16)v;
                }
            }
        }
        // ---- stage B tile TRANSPOSED: BsT[n][k], 32 k-rows x 32 n
        {
            const int r  = t >> 3;                 // k row 0..31
            const int c0 = (t & 7) << 2;           // 4 cols
            const int gk = kb + r;
            if (fullN && fullK) {
                const float4 v = *(const float4*)(Bm + (long)gk * N + n0 + c0);
                BsT[(c0 + 0) * BT_STRIDE + r] = (_Float16)v.x;
                BsT[(c0 + 1) * BT_STRIDE + r] = (_Float16)v.y;
                BsT[(c0 + 2) * BT_STRIDE + r] = (_Float16)v.z;
                BsT[(c0 + 3) * BT_STRIDE + r] = (_Float16)v.w;
            } else {
                const bool kok = (gk < K);
                const int  gkc = kok ? gk : (K - 1);
                const float* bp = Bm + (long)gkc * N;
                #pragma unroll
                for (int j = 0; j < 4; ++j) {
                    const int n  = n0 + c0 + j;
                    const int nc = (n < N) ? n : (N - 1);
                    float v = bp[nc];
                    v = (kok && (n < N)) ? v : 0.0f;
                    BsT[(c0 + j) * BT_STRIDE + r] = (_Float16)v;
                }
            }
        }
        __syncthreads();

        // ---- build fragments per ISA layout (CDNA5 ISA §7.12.2)
        // A 16x32 f16: lane's 16 halves are k = 8*hi+0..7 and 16+8*hi+0..7,
        // contiguous in the As row -> two ds_load_b128.
        const int mrow = (wave << 4) + (lane & 15);
        const int hi   = (lane & 16) ? 1 : 0;
        const int bcol = lane & 15;

        union { v16h v; v8h h8[2]; } af, bf0, bf1;
        {
            const _Float16* ar = &As[mrow * AS_STRIDE + (hi << 3)];
            af.h8[0] = *(const v8h*)(ar);          // k = 8hi .. 8hi+7
            af.h8[1] = *(const v8h*)(ar + 16);     // k = 16+8hi .. 16+8hi+7
        }
        // B 32x16 f16 per subtile: lane's 16 halves are k = 8*hi+0..7 and
        // 16+8*hi+0..7 for column n -> in BsT[n][*] -> two ds_load_b128 each.
        {
            const _Float16* b0 = &BsT[bcol * BT_STRIDE + (hi << 3)];
            bf0.h8[0] = *(const v8h*)(b0);
            bf0.h8[1] = *(const v8h*)(b0 + 16);
            const _Float16* b1 = &BsT[(bcol + 16) * BT_STRIDE + (hi << 3)];
            bf1.h8[0] = *(const v8h*)(b1);
            bf1.h8[1] = *(const v8h*)(b1 + 16);
        }

        acc0 = __builtin_amdgcn_wmma_f32_16x16x32_f16(false, af.v, false, bf0.v,
                                                      (short)0, acc0, false, false);
        acc1 = __builtin_amdgcn_wmma_f32_16x16x32_f16(false, af.v, false, bf1.v,
                                                      (short)0, acc1, false, false);
        __syncthreads();
    }

    // ---- epilogue: bias + optional ReLU, strided store (C/D layout: VGPR v
    // holds M = v (lanes 0-15) / v+8 (lanes 16-31), N = lane&15)
    const int bcol = lane & 15;
    const int hi8  = (lane & 16) ? 8 : 0;
    #pragma unroll
    for (int v = 0; v < 8; ++v) {
        const long m = (long)m0 + (wave << 4) + v + hi8;
        if (m >= M) continue;
        const int n1 = n0 + bcol;
        if (n1 < N) {
            float val = acc0[v] + bias[n1];
            if (relu) val = fmaxf(val, 0.0f);
            C[m * (long)ldc + n1] = val;
        }
        const int n2 = n0 + 16 + bcol;
        if (n2 < N) {
            float val = acc1[v] + bias[n2];
            if (relu) val = fmaxf(val, 0.0f);
            C[m * (long)ldc + n2] = val;
        }
    }
}

// ---------------------------------------------------------------------------
// Support kernels
// ---------------------------------------------------------------------------
__global__ void copy_f32_v4(const float4* __restrict__ src, float4* __restrict__ dst,
                            long n4)
{
    const long i = (long)blockIdx.x * blockDim.x + threadIdx.x;
    if (i < n4) dst[i] = src[i];
}

__global__ void fill_zero(float* __restrict__ p, long n)
{
    const long i = (long)blockIdx.x * blockDim.x + threadIdx.x;
    if (i < n) p[i] = 0.0f;
}

// agg[dst] += x[src]; one wave per edge, lane = channel
__global__ void edge_scatter(const float* __restrict__ x, const int* __restrict__ ei,
                             float* __restrict__ agg, int E, int F)
{
    const int wid  = (int)(((long)blockIdx.x * blockDim.x + threadIdx.x) >> 5);
    const int lane = threadIdx.x & 31;
    if (wid >= E) return;
    const long s = ei[wid];
    const long d = ei[(long)E + wid];
    for (int f = lane; f < F; f += 32)
        atomicAdd(&agg[d * F + f], x[s * F + f]);
}

// per-channel sum / sumsq over rows (DIM=32), partials via LDS + atomics
__global__ void bn_stats(const float* __restrict__ h, float* __restrict__ stats, int n)
{
    const int c = threadIdx.x & 31;
    float s = 0.0f, q = 0.0f;
    for (long r = (long)blockIdx.x * 8 + (threadIdx.x >> 5); r < n;
         r += (long)gridDim.x * 8) {
        const float v = h[r * DIM + c];
        s += v; q += v * v;
    }
    __shared__ float ls[256], lq[256];
    ls[threadIdx.x] = s; lq[threadIdx.x] = q;
    __syncthreads();
    if (threadIdx.x < 32) {
        for (int i = 1; i < 8; ++i) { s += ls[c + 32 * i]; q += lq[c + 32 * i]; }
        atomicAdd(&stats[c], s);
        atomicAdd(&stats[32 + c], q);
    }
}

__global__ void bn_apply(float* __restrict__ h, const float* __restrict__ stats,
                         const float* __restrict__ gamma, const float* __restrict__ beta,
                         long rows)
{
    const long i = (long)blockIdx.x * blockDim.x + threadIdx.x;
    if (i >= rows * DIM) return;
    const int c = (int)(i & 31);
    const float invn = 1.0f / (float)rows;
    const float mean = stats[c] * invn;
    const float var  = stats[32 + c] * invn - mean * mean;
    const float sc   = gamma[c] * rsqrtf(var + BN_EPS);
    h[i] = (h[i] - mean) * sc + beta[c];
}

// global_add_pool: wave per node, lane = channel
__global__ void pool_nodes(const float* __restrict__ h, const int* __restrict__ batch,
                           float* __restrict__ pooled, int n)
{
    const int wid  = (int)(((long)blockIdx.x * blockDim.x + threadIdx.x) >> 5);
    const int lane = threadIdx.x & 31;
    if (wid >= n) return;
    const long b = batch[wid];
    atomicAdd(&pooled[b * DIM + lane], h[(long)wid * DIM + lane]);
}

// row-wise L2 normalize, one block per row
__global__ void l2norm_rows(const float* __restrict__ in, float* __restrict__ out, int cols)
{
    const long row = blockIdx.x;
    const float* r = in + row * cols;
    float s = 0.0f;
    for (int c = threadIdx.x; c < cols; c += blockDim.x) { const float v = r[c]; s += v * v; }
    __shared__ float ls[256];
    ls[threadIdx.x] = s;
    __syncthreads();
    for (int off = 128; off; off >>= 1) {
        if (threadIdx.x < off) ls[threadIdx.x] += ls[threadIdx.x + off];
        __syncthreads();
    }
    const float inv = 1.0f / fmaxf(sqrtf(ls[0]), 1e-12f);
    float* o = out + row * cols;
    for (int c = threadIdx.x; c < cols; c += blockDim.x) o[c] = r[c] * inv;
}

// ---------------------------------------------------------------------------
// Host-side orchestration
// ---------------------------------------------------------------------------
static inline int cdiv(long a, long b) { return (int)((a + b - 1) / b); }

static void run_gemm(const float* A, const float* B, const float* bias, float* C,
                     int M, int N, int K, int lda, int ldc, int relu, hipStream_t s)
{
    dim3 grid(cdiv(M, 128), cdiv(N, 32));
    gemm_wmma_f16<<<grid, 256, 0, s>>>(A, B, bias, C, M, N, K, lda, ldc, relu);
}

extern "C" void kernel_launch(void* const* d_in, const int* in_sizes, int n_in,
                              void* d_out, int out_size, void* d_ws, size_t ws_size,
                              hipStream_t stream)
{
    (void)in_sizes; (void)n_in; (void)out_size; (void)ws_size;

    const float* x1     = (const float*)d_in[0];
    const int*   ei1    = (const int*)  d_in[1];
    const int*   batch1 = (const int*)  d_in[2];
    const float* x2     = (const float*)d_in[3];
    const int*   ei2    = (const int*)  d_in[4];
    const int*   batch2 = (const int*)  d_in[5];
    const float* cell   = (const float*)d_in[6];
    const float* g1_w1  = (const float*)d_in[7];
    const float* g1_b1  = (const float*)d_in[8];
    const float* g1_w2  = (const float*)d_in[9];
    const float* g1_b2  = (const float*)d_in[10];
    const float* gw1    = (const float*)d_in[11];
    const float* gb1    = (const float*)d_in[12];
    const float* gw2    = (const float*)d_in[13];
    const float* gb2    = (const float*)d_in[14];
    const float* bn_g   = (const float*)d_in[15];
    const float* bn_b   = (const float*)d_in[16];
    const float* fcd_w  = (const float*)d_in[17];
    const float* fcd_b  = (const float*)d_in[18];
    const float* red_w1 = (const float*)d_in[19];
    const float* red_b1 = (const float*)d_in[20];
    const float* red_w2 = (const float*)d_in[21];
    const float* red_b2 = (const float*)d_in[22];
    const float* red_w3 = (const float*)d_in[23];
    const float* red_b3 = (const float*)d_in[24];
    const float* fc1_w  = (const float*)d_in[25];
    const float* fc1_b  = (const float*)d_in[26];
    const float* fc2_w  = (const float*)d_in[27];
    const float* fc2_b  = (const float*)d_in[28];
    const float* out_w  = (const float*)d_in[29];
    const float* out_b  = (const float*)d_in[30];

    float* ws = (float*)d_ws;

    // workspace layout (floats); cell/head regions alias freed graph regions
    const long AGG   = 0;                        // 250000*78 = 19,500,000
    const long H1    = 19500032;                 // 250000*32 = 8,000,000
    const long H2    = H1 + 8000000;
    const long STATS = H2 + 8000000;             // 64
    const long POOL  = STATS + 64;               // 8192*32 = 262,144
    const long XC    = POOL + 262144;            // 8192*384 = 3,145,728
    const long CELLN = 0;                        // alias AGG (after branches)
    const long C1    = 8000000;                  // alias AGG
    const long C2    = 12500000;                 // alias AGG
    const long HH1   = H1;                       // alias H1/H2
    const long HH2   = H1 + 8388608;             // alias H2

    const long nodeElems = (long)NN * DIM;

    auto bn_layer = [&](float* h, const float* g, const float* b) {
        fill_zero<<<1, 64, 0, stream>>>(ws + STATS, 64);
        bn_stats<<<1024, 256, 0, stream>>>(h, ws + STATS, NN);
        bn_apply<<<cdiv(nodeElems, 256), 256, 0, stream>>>(h, ws + STATS, g, b, NN);
    };

    auto run_branch = [&](const float* x, const int* ei, const int* batch, float* xc_dst) {
        // ---- GIN layer 1 (78 -> 32 -> 32)
        copy_f32_v4<<<cdiv((long)NN * FXD / 4, 256), 256, 0, stream>>>(
            (const float4*)x, (float4*)(ws + AGG), (long)NN * FXD / 4);
        edge_scatter<<<cdiv(EE, 8), 256, 0, stream>>>(x, ei, ws + AGG, EE, FXD);
        run_gemm(ws + AGG, g1_w1, g1_b1, ws + H1, NN, DIM, FXD, FXD, DIM, 1, stream);
        run_gemm(ws + H1, g1_w2, g1_b2, ws + H2, NN, DIM, DIM, DIM, DIM, 1, stream);
        bn_layer(ws + H2, bn_g + 0 * DIM, bn_b + 0 * DIM);

        // ---- GIN layers 2..5 (32 -> 32 -> 32)
        for (int i = 0; i < 4; ++i) {
            copy_f32_v4<<<cdiv(nodeElems / 4, 256), 256, 0, stream>>>(
                (const float4*)(ws + H2), (float4*)(ws + AGG), nodeElems / 4);
            edge_scatter<<<cdiv(EE, 8), 256, 0, stream>>>(ws + H2, ei, ws + AGG, EE, DIM);
            run_gemm(ws + AGG, gw1 + (long)i * DIM * DIM, gb1 + (long)i * DIM,
                     ws + H1, NN, DIM, DIM, DIM, DIM, 1, stream);
            run_gemm(ws + H1, gw2 + (long)i * DIM * DIM, gb2 + (long)i * DIM,
                     ws + H2, NN, DIM, DIM, DIM, DIM, 1, stream);
            bn_layer(ws + H2, bn_g + (long)(i + 1) * DIM, bn_b + (long)(i + 1) * DIM);
        }

        // ---- global_add_pool + fcd (32 -> 128), written into concat buffer
        fill_zero<<<cdiv((long)BB * DIM, 256), 256, 0, stream>>>(ws + POOL, (long)BB * DIM);
        pool_nodes<<<cdiv(NN, 8), 256, 0, stream>>>(ws + H2, batch, ws + POOL, NN);
        run_gemm(ws + POOL, fcd_w, fcd_b, xc_dst, BB, 128, DIM, DIM, 384, 1, stream);
    };

    // drug branches -> xc[:, 0:128] and xc[:, 128:256]
    run_branch(x1, ei1, batch1, ws + XC + 0);
    run_branch(x2, ei2, batch2, ws + XC + 128);

    // cell branch -> xc[:, 256:384]
    l2norm_rows<<<BB, 256, 0, stream>>>(cell, ws + CELLN, 954);
    run_gemm(ws + CELLN, red_w1, red_b1, ws + C1, BB, 512, 954, 954, 512, 1, stream);
    run_gemm(ws + C1, red_w2, red_b2, ws + C2, BB, 256, 512, 512, 256, 1, stream);
    run_gemm(ws + C2, red_w3, red_b3, ws + XC + 256, BB, 128, 256, 256, 384, 0, stream);

    // head
    run_gemm(ws + XC, fc1_w, fc1_b, ws + HH1, BB, 1024, 384, 384, 1024, 1, stream);
    run_gemm(ws + HH1, fc2_w, fc2_b, ws + HH2, BB, 256, 1024, 1024, 256, 1, stream);
    run_gemm(ws + HH2, out_w, out_b, (float*)d_out, BB, 2, 256, 256, 2, 0, stream);
}
